// ResidualVectorQuantizer_2774548873907
// MI455X (gfx1250) — compile-verified, matrix operand order checked by reading, not run on hardware
//
#include <hip/hip_runtime.h>

// ---------------------------------------------------------------------------
// Residual Vector Quantizer for MI455X (gfx1250, wave32, WMMA)
//   inputs:    (16, 4096, 512) f32  -> N = 65536 rows, D = 512
//   codebooks: (4, 1024, 512)  f32  -> L = 4 levels, K = 1024 codes
// Outputs (concatenated in d_out, float):
//   [0 .. 33554431]  final_ste  = inputs - residual_final
//   [33554432]       q_loss/4   (== e_loss numerically)
//   [33554433]       e_loss/4
//   [33554434 ..]    indices (4, 16, 4096) written as float values
// ---------------------------------------------------------------------------

typedef __attribute__((ext_vector_type(16))) __bf16 v16bf;
typedef __attribute__((ext_vector_type(8)))  float  v8f;

#define D_DIM   512
#define KCODES  1024
#define LVLS    4
#define RPB     64        // rows per block
#define NROWS   65536
#define O_QL    33554432ull
#define O_EL    33554433ull
#define O_IDX   33554434ull

static __device__ __forceinline__ unsigned short f2bf(float f) {
  unsigned u = __float_as_uint(f);
  u += 0x7fffu + ((u >> 16) & 1u);     // round-to-nearest-even
  return (unsigned short)(u >> 16);
}
static __device__ __forceinline__ float bf2f(unsigned short h) {
  return __uint_as_float(((unsigned)h) << 16);
}

struct alignas(16) Frag32 { uint4 a, b; };

// A-fragment from LDS: two 16B segments 16 elements apart (K pattern of
// 16-bit A 16x32: lanes<16 hold K 0..7 & 16..23, lanes>=16 hold 8..15 & 24..31)
static __device__ __forceinline__ v16bf ld_frag_lds(const unsigned short* p) {
  Frag32 f;
  f.a = *reinterpret_cast<const uint4*>(p);
  f.b = *reinterpret_cast<const uint4*>(p + 16);
  return __builtin_bit_cast(v16bf, f);
}
// B-fragment from the pre-swizzled global array: seg0 at p[0], seg1 at p[32]
static __device__ __forceinline__ v16bf ld_frag_g(const uint4* p) {
  Frag32 f;
  f.a = p[0];
  f.b = p[32];
  return __builtin_bit_cast(v16bf, f);
}

// ---------------------------------------------------------------------------
// Codebook row norms ||c_k||^2  (4096 rows)
// ---------------------------------------------------------------------------
__global__ void rvq_norms(const float* __restrict__ cb, float* __restrict__ norms) {
  int id = blockIdx.x * blockDim.x + threadIdx.x;
  if (id >= LVLS * KCODES) return;
  const float4* row = reinterpret_cast<const float4*>(cb) + (size_t)id * (D_DIM / 4);
  float acc = 0.f;
  for (int j = 0; j < D_DIM / 4; ++j) {
    float4 v = row[j];
    acc += v.x * v.x + v.y * v.y + v.z * v.z + v.w * v.w;
  }
  norms[id] = acc;
}

// ---------------------------------------------------------------------------
// Repack codebooks into bf16 hi/lo B-fragments, already in WMMA lane order.
// Flat uint4 index q = ((((lvl*64+kt)*16+ch)*2+seg)*32)+lane
//   code = kt*16 + (lane&15)
//   d0   = ch*32 + (lane>=16 ? 8 : 0) + seg*16, elements d0..d0+7
// ---------------------------------------------------------------------------
__global__ void rvq_make_frags(const float* __restrict__ cb,
                               uint4* __restrict__ fhi, uint4* __restrict__ flo) {
  unsigned q = blockIdx.x * blockDim.x + threadIdx.x;   // 0..262143
  unsigned lane = q & 31u;
  unsigned seg  = (q >> 5) & 1u;
  unsigned ch   = (q >> 6) & 15u;
  unsigned kt   = (q >> 10) & 63u;
  unsigned lvl  = q >> 16;
  unsigned code = kt * 16u + (lane & 15u);
  unsigned d0   = ch * 32u + ((lane >> 4) << 3) + seg * 16u;
  const float* src = cb + ((size_t)(lvl * KCODES + code)) * D_DIM + d0;
  unsigned hi[8], lo[8];
#pragma unroll
  for (int j = 0; j < 8; ++j) {
    float v = src[j];
    unsigned short h = f2bf(v);
    unsigned short l = f2bf(v - bf2f(h));
    hi[j] = h; lo[j] = l;
  }
  fhi[q] = make_uint4(hi[0] | (hi[1] << 16), hi[2] | (hi[3] << 16),
                      hi[4] | (hi[5] << 16), hi[6] | (hi[7] << 16));
  flo[q] = make_uint4(lo[0] | (lo[1] << 16), lo[2] | (lo[3] << 16),
                      lo[4] | (lo[5] << 16), lo[6] | (lo[7] << 16));
}

// ---------------------------------------------------------------------------
// Main fused kernel: 4 waves / block, 64 rows / block, residual kept in LDS
// as m = -2*r in bf16 hi/lo across all 4 levels.
// score(n,k) = ||c_k||^2 + m_n . c_k  via bf16x3 WMMA (hi*hi + hi*lo + lo*hi)
// ---------------------------------------------------------------------------
__global__ __launch_bounds__(128) void rvq_main(
    const float* __restrict__ in, const float* __restrict__ cb,
    const float* __restrict__ norms, const uint4* __restrict__ fhi,
    const uint4* __restrict__ flo, float* __restrict__ out,
    float* __restrict__ partials) {
  __shared__ unsigned short Ahi[RPB * D_DIM];   // 64 KB
  __shared__ unsigned short Alo[RPB * D_DIM];   // 64 KB
  __shared__ int   idx_s[RPB];
  __shared__ float red_s[128];

  const int t    = threadIdx.x;
  const int lane = t & 31;
  const int wave = t >> 5;        // 0..3, owns matrix rows wave*16..wave*16+15
  const int half = lane >> 4;
  const int lrow = lane & 15;
  const size_t rowbase = (size_t)blockIdx.x * RPB;
  const float4* in4  = reinterpret_cast<const float4*>(in);
  const float4* cb4  = reinterpret_cast<const float4*>(cb);
  float4*       out4 = reinterpret_cast<float4*>(out);
  float*        idxout = out + O_IDX;

  // ---- stage m = -2*x as bf16 hi/lo into LDS ----
#pragma unroll 4
  for (int r = 0; r < RPB; ++r) {
    float4 x = in4[(rowbase + r) * (D_DIM / 4) + t];
    float m0 = -2.f * x.x, m1 = -2.f * x.y, m2 = -2.f * x.z, m3 = -2.f * x.w;
    int e = r * D_DIM + t * 4;
    unsigned short h0 = f2bf(m0), h1 = f2bf(m1), h2 = f2bf(m2), h3 = f2bf(m3);
    *reinterpret_cast<ushort4*>(&Ahi[e]) = make_ushort4(h0, h1, h2, h3);
    *reinterpret_cast<ushort4*>(&Alo[e]) = make_ushort4(
        f2bf(m0 - bf2f(h0)), f2bf(m1 - bf2f(h1)),
        f2bf(m2 - bf2f(h2)), f2bf(m3 - bf2f(h3)));
  }
  __syncthreads();

  float lossAcc = 0.f;   // accumulates sum(m_new^2) = 4*sum(r_new^2)

#pragma unroll 1
  for (int lvl = 0; lvl < LVLS; ++lvl) {
    float bestS[8];
    int   bestI[8];
#pragma unroll
    for (int v = 0; v < 8; ++v) { bestS[v] = 3.402823466e38f; bestI[v] = 0; }
    const float* nrm_lvl = norms + lvl * KCODES;

#pragma unroll 1
    for (int ktg = 0; ktg < 16; ++ktg) {       // groups of 4 k-tiles
      v8f C[4];
#pragma unroll
      for (int g = 0; g < 4; ++g) {
        float nv = nrm_lvl[(ktg * 4 + g) * 16 + lrow];
#pragma unroll
        for (int e = 0; e < 8; ++e) C[g][e] = nv;
      }
#pragma unroll 2
      for (int ch = 0; ch < 16; ++ch) {        // D = 16 chunks of 32
        const int abase = (wave * 16 + lrow) * D_DIM + ch * 32 + half * 8;
        v16bf ah = ld_frag_lds(&Ahi[abase]);
        v16bf al = ld_frag_lds(&Alo[abase]);
#pragma unroll
        for (int g = 0; g < 4; ++g) {
          const size_t fb =
              ((((size_t)lvl * 64 + (ktg * 4 + g)) * 16 + ch) * 2) * 32 + lane;
          v16bf bh = ld_frag_g(fhi + fb);
          v16bf bl = ld_frag_g(flo + fb);
          C[g] = __builtin_amdgcn_wmma_f32_16x16x32_bf16(
              false, ah, false, bh, (short)0, C[g], false, false);
          C[g] = __builtin_amdgcn_wmma_f32_16x16x32_bf16(
              false, ah, false, bl, (short)0, C[g], false, false);
          C[g] = __builtin_amdgcn_wmma_f32_16x16x32_bf16(
              false, al, false, bh, (short)0, C[g], false, false);
        }
      }
      // per-lane running argmin (columns visited in increasing index order,
      // strict < keeps the lowest index on ties, matching jnp.argmin)
#pragma unroll
      for (int g = 0; g < 4; ++g) {
        const int colIdx = (ktg * 4 + g) * 16 + lrow;
#pragma unroll
        for (int v = 0; v < 8; ++v) {
          float s = C[g][v];
          if (s < bestS[v]) { bestS[v] = s; bestI[v] = colIdx; }
        }
      }
    }
    // cross-lane reduce within each 16-lane group (C layout: lanes 0-15 row v,
    // lanes 16-31 row v+8); lexicographic (score, index) min
#pragma unroll
    for (int v = 0; v < 8; ++v) {
      float s = bestS[v];
      int   i = bestI[v];
#pragma unroll
      for (int m = 1; m <= 8; m <<= 1) {
        float os = __shfl_xor(s, m, 32);
        int   oi = __shfl_xor(i, m, 32);
        if (os < s || (os == s && oi < i)) { s = os; i = oi; }
      }
      if (lrow == v) idx_s[wave * 16 + v + half * 8] = i;
    }
    __syncthreads();

    if (t < RPB)
      idxout[(size_t)lvl * NROWS + rowbase + t] = (float)idx_s[t];

    // residual update: m += 2*c[idx]; loss += m_new^2; restage hi/lo
#pragma unroll 2
    for (int r = 0; r < RPB; ++r) {
      const int code = idx_s[r];
      float4 c = cb4[((size_t)lvl * KCODES + code) * (D_DIM / 4) + t];
      const int e = r * D_DIM + t * 4;
      ushort4 h = *reinterpret_cast<const ushort4*>(&Ahi[e]);
      ushort4 l = *reinterpret_cast<const ushort4*>(&Alo[e]);
      float m0 = bf2f(h.x) + bf2f(l.x) + 2.f * c.x;
      float m1 = bf2f(h.y) + bf2f(l.y) + 2.f * c.y;
      float m2 = bf2f(h.z) + bf2f(l.z) + 2.f * c.z;
      float m3 = bf2f(h.w) + bf2f(l.w) + 2.f * c.w;
      lossAcc += m0 * m0 + m1 * m1 + m2 * m2 + m3 * m3;
      unsigned short h0 = f2bf(m0), h1 = f2bf(m1), h2 = f2bf(m2), h3 = f2bf(m3);
      *reinterpret_cast<ushort4*>(&Ahi[e]) = make_ushort4(h0, h1, h2, h3);
      *reinterpret_cast<ushort4*>(&Alo[e]) = make_ushort4(
          f2bf(m0 - bf2f(h0)), f2bf(m1 - bf2f(h1)),
          f2bf(m2 - bf2f(h2)), f2bf(m3 - bf2f(h3)));
    }
    __syncthreads();
  }

  // final_ste = x - r_final = x + m/2
#pragma unroll 4
  for (int r = 0; r < RPB; ++r) {
    float4 x = in4[(rowbase + r) * (D_DIM / 4) + t];
    const int e = r * D_DIM + t * 4;
    ushort4 h = *reinterpret_cast<const ushort4*>(&Ahi[e]);
    ushort4 l = *reinterpret_cast<const ushort4*>(&Alo[e]);
    float4 o;
    o.x = x.x + 0.5f * (bf2f(h.x) + bf2f(l.x));
    o.y = x.y + 0.5f * (bf2f(h.y) + bf2f(l.y));
    o.z = x.z + 0.5f * (bf2f(h.z) + bf2f(l.z));
    o.w = x.w + 0.5f * (bf2f(h.w) + bf2f(l.w));
    out4[(rowbase + r) * (D_DIM / 4) + t] = o;
  }

  // deterministic block loss reduction
  red_s[t] = lossAcc;
  __syncthreads();
#pragma unroll
  for (int s = 64; s > 0; s >>= 1) {
    if (t < s) red_s[t] += red_s[t + s];
    __syncthreads();
  }
  if (t == 0) partials[blockIdx.x] = red_s[0];
}

// ---------------------------------------------------------------------------
// Deterministic final loss reduction:
//   loss = sum(m^2) / 4 (m->r) / (N*D) (mean) / 4 (levels) = sum / 2^29
// ---------------------------------------------------------------------------
__global__ void rvq_finalize(const float* __restrict__ partials, float* __restrict__ out) {
  __shared__ float red[256];
  int t = threadIdx.x;
  float s = 0.f;
#pragma unroll
  for (int j = 0; j < 4; ++j) s += partials[t * 4 + j];
  red[t] = s;
  __syncthreads();
#pragma unroll
  for (int k = 128; k > 0; k >>= 1) {
    if (t < k) red[t] += red[t + k];
    __syncthreads();
  }
  if (t == 0) {
    float v = red[0] * (1.0f / 536870912.0f);
    out[O_QL] = v;
    out[O_EL] = v;
  }
}

// ---------------------------------------------------------------------------
extern "C" void kernel_launch(void* const* d_in, const int* in_sizes, int n_in,
                              void* d_out, int out_size, void* d_ws, size_t ws_size,
                              hipStream_t stream) {
  (void)in_sizes; (void)n_in; (void)out_size; (void)ws_size;
  const float* in = (const float*)d_in[0];
  const float* cb = (const float*)d_in[1];
  float* out = (float*)d_out;
  char* ws = (char*)d_ws;
  // workspace layout: norms 16KB | partials 4KB | pad | frag_hi 4MB | frag_lo 4MB
  float* norms    = (float*)ws;
  float* partials = (float*)(ws + 16384);
  uint4* fhi      = (uint4*)(ws + 32768);
  uint4* flo      = (uint4*)(ws + 32768 + 4194304);

  rvq_norms<<<16, 256, 0, stream>>>(cb, norms);
  rvq_make_frags<<<1024, 256, 0, stream>>>(cb, fhi, flo);
  rvq_main<<<NROWS / RPB, 128, 0, stream>>>(in, cb, norms, fhi, flo, out, partials);
  rvq_finalize<<<1, 256, 0, stream>>>(partials, out);
}
